// GNN_57681410785844
// MI455X (gfx1250) — matrix operand order repaired and checked by reading, not compiled, not run on hardware
//
#include <hip/hip_runtime.h>

// ---------------------------------------------------------------------------
// CDNA5 (gfx1250) fused GNN: dense all-pairs message passing via bf16 WMMA.
// Wave-local LDS dataflow: 2 barriers/tile, hid fragments built in registers,
// native __bf16 conversions (hardware packed cvt).
// ---------------------------------------------------------------------------

typedef __attribute__((ext_vector_type(16))) __bf16       v16bf;
typedef __attribute__((ext_vector_type(8)))  float        v8f;
typedef __attribute__((ext_vector_type(8)))  unsigned int v8u;

#define SILU_GAIN 1.6768f
#define LN_EPS    1e-6f
#define INV_DEG   (1.0f / 767.0f)
#define INV_SQRT2 0.70710678118654752f

static constexpr int NN   = 768;   // nodes
static constexpr int NB   = 48;    // 16-node sender blocks
static constexpr int JS   = 16;    // receiver slices
static constexpr int TPS  = 3;     // tiles per slice (48/16)
static constexpr int HIDP = 96;    // msg hidden, padded 91->96
static constexpr int HIDR = 91;

// ---- LDS layout (bytes) for the edge kernel (dynamic shared) --------------
static constexpr int O_POSS  = 0;        // 48 f32
static constexpr int O_POSR  = 192;      // 48 f32
static constexpr int O_SIG   = 384;      // 32 f32  (1/sigma^2)
static constexpr int O_WEDGE = 512;      // 128 f32
static constexpr int O_BEDGE = 1024;     // 32 f32
static constexpr int O_LNS   = 1152;     // 32 f32
static constexpr int O_LNB   = 1280;     // 32 f32
static constexpr int O_MB1   = 1408;     // 64 f32 (msg final bias)
static constexpr int O_AROWS = 1664;     // 16*96 f32
static constexpr int O_BROWS = 7808;     // 16*96 f32
static constexpr int O_HN    = 13952;    // 256*32 f32   LN(h) per edge
static constexpr int O_WENVT = 46720;    // 32*32 bf16 (transposed)
static constexpr int O_WGT   = 48768;    // 64*32 bf16 (transposed)
static constexpr int O_W2T   = 52864;    // 64*96 bf16 (transposed, padded)
static constexpr int O_RBFA  = 65152;    // 256*32 bf16 (aliased: e_embed A-matrix)
static constexpr int SMEM_BYTES = 81536; // 79.6 KB -> 4 workgroups / 320KB WGP

// ---------------------------------------------------------------------------

__device__ __forceinline__ unsigned short f2bf(float f) {
  return __builtin_bit_cast(unsigned short, static_cast<__bf16>(f));
}
// packed pair -> one dword (backend can use packed bf16 convert)
__device__ __forceinline__ unsigned f2bf2(float x0, float x1) {
  return (unsigned)f2bf(x0) | ((unsigned)f2bf(x1) << 16);
}
__device__ __forceinline__ float silu_g(float x) {
  return x * __builtin_amdgcn_rcpf(1.0f + __expf(-x)) * SILU_GAIN;
}

// A-matrix fragment (16x32 bf16), ISA 7.12.2:
// lane<16: V0-3 = K[kb..kb+7], V4-7 = K[16+kb..16+kb+7], kb = (lane>>4)*8
__device__ __forceinline__ v16bf ldsAfrag(const unsigned short* buf, int stride,
                                          int row0, int k0) {
  const int lane = threadIdx.x & 31;
  const int lo = lane & 15, hi = lane >> 4;
  const unsigned short* p = buf + (row0 + lo) * stride + k0 + hi * 8;
  uint4 c0 = *(const uint4*)(p);
  uint4 c1 = *(const uint4*)(p + 16);
  v8u pk;
  pk[0] = c0.x; pk[1] = c0.y; pk[2] = c0.z; pk[3] = c0.w;
  pk[4] = c1.x; pk[5] = c1.y; pk[6] = c1.z; pk[7] = c1.w;
  return __builtin_bit_cast(v16bf, pk);
}

// B-matrix fragment (32x16 bf16) from N-major (transposed) weights:
// lane: N = lane&15, V0-3 = K[kb..kb+7], V4-7 = K[16+kb..16+kb+7], kb = (lane>>4)*8
__device__ __forceinline__ v16bf ldsBfrag(const unsigned short* wt, int stride,
                                          int n0, int k0) {
  const int lane = threadIdx.x & 31;
  const int lo = lane & 15, hi = lane >> 4;
  const unsigned short* p = wt + (n0 + lo) * stride + k0 + hi * 8;
  uint4 c0 = *(const uint4*)(p);
  uint4 c1 = *(const uint4*)(p + 16);
  v8u pk;
  pk[0] = c0.x; pk[1] = c0.y; pk[2] = c0.z; pk[3] = c0.w;
  pk[4] = c1.x; pk[5] = c1.y; pk[6] = c1.z; pk[7] = c1.w;
  return __builtin_bit_cast(v16bf, pk);
}

// hid A-fragment built directly in registers: silu(A[sender m] + B[receiver lo])
__device__ __forceinline__ v16bf hidAfrag(const float* sA, const float* sB,
                                          int m, int k0) {
  const int lane = threadIdx.x & 31;
  const int lo = lane & 15, kb = (lane >> 4) * 8;
  const float* arow = sA + m * HIDP;        // uniform per wave (broadcast)
  const float* brow = sB + lo * HIDP;
  v8u pk;
#pragma unroll
  for (int h = 0; h < 2; ++h) {
    const int tb = k0 + 16 * h + kb;
#pragma unroll
    for (int p = 0; p < 4; ++p) {
      float x0 = silu_g(arow[tb + 2 * p]     + brow[tb + 2 * p]);
      float x1 = silu_g(arow[tb + 2 * p + 1] + brow[tb + 2 * p + 1]);
      pk[h * 4 + p] = f2bf2(x0, x1);
    }
  }
  return __builtin_bit_cast(v16bf, pk);
}

__device__ __forceinline__ v8f wmma_bf16(v16bf a, v16bf b, v8f c) {
  return __builtin_amdgcn_wmma_f32_16x16x32_bf16(false, a, false, b,
                                                 (short)0, c, false, false);
}

// ---------------------------------------------------------------------------
// K0: node embedding lookup
__global__ void gnn_embed_k(const int* __restrict__ charges,
                            const float* __restrict__ embed,
                            float* __restrict__ H0) {
  int idx = blockIdx.x * blockDim.x + threadIdx.x;
  if (idx < NN * 64) {
    int n = idx >> 6, k = idx & 63;
    H0[idx] = embed[charges[n] * 64 + k];
  }
}

// K1: per-layer precompute  A = H*W1_top + b0 (padded), B = H*W1_bot
__global__ void gnn_precab_k(const float* __restrict__ H,
                             const float* __restrict__ W0,   // [128][91]
                             const float* __restrict__ b0,   // [91]
                             float* __restrict__ Abuf,       // [768][96]
                             float* __restrict__ Bbuf) {     // [768][96]
  int idx = blockIdx.x * blockDim.x + threadIdx.x;
  if (idx >= NN * HIDP) return;
  int n = idx / HIDP, t = idx % HIDP;
  float a = 0.f, b = 0.f;
  if (t < HIDR) {
    a = b0[t];
    const float* h = H + n * 64;
    for (int k = 0; k < 64; ++k) {
      a += h[k] * W0[k * HIDR + t];
      b += h[k] * W0[(64 + k) * HIDR + t];
    }
  }
  Abuf[idx] = a;
  Bbuf[idx] = b;
}

// ---------------------------------------------------------------------------
// K2: fused edge kernel. grid = (48 sender blocks, 16 receiver slices),
// 256 threads = 8 waves. All per-edge GEMMs via bf16 WMMA, wave-local LDS.
__global__ __launch_bounds__(256)
void gnn_edge_k(const float* __restrict__ pos,      // [768][3]
                const float* __restrict__ Abuf,     // [768][96]
                const float* __restrict__ Bbuf,     // [768][96]
                const float* __restrict__ sigma,    // [32]
                const float* __restrict__ Wenv,     // [32][32]
                const float* __restrict__ Wedge,    // [4][32]
                const float* __restrict__ bedge,    // [32]
                const float* __restrict__ lns,      // [32]
                const float* __restrict__ lnb,      // [32]
                const float* __restrict__ Wgate,    // [32][64]
                const float* __restrict__ msgW2,    // [91][64]
                const float* __restrict__ msgB1,    // [64]
                float* __restrict__ msg_part) {     // [JS][768][64]
  extern __shared__ __attribute__((aligned(128))) char smem[];
  float* sPosS  = (float*)(smem + O_POSS);
  float* sPosR  = (float*)(smem + O_POSR);
  float* sSig   = (float*)(smem + O_SIG);
  float* sWedge = (float*)(smem + O_WEDGE);
  float* sBedge = (float*)(smem + O_BEDGE);
  float* sLns   = (float*)(smem + O_LNS);
  float* sLnb   = (float*)(smem + O_LNB);
  float* sMb1   = (float*)(smem + O_MB1);
  float* sArows = (float*)(smem + O_AROWS);
  float* sBrows = (float*)(smem + O_BROWS);
  float* sHn    = (float*)(smem + O_HN);
  unsigned short* sWenvT = (unsigned short*)(smem + O_WENVT);
  unsigned short* sWgT   = (unsigned short*)(smem + O_WGT);
  unsigned short* sW2T   = (unsigned short*)(smem + O_W2T);
  unsigned short* sRbfA  = (unsigned short*)(smem + O_RBFA);
  unsigned short* sEA    = sRbfA;     // aliased; wave-local rows, safe

  const int tid   = threadIdx.x;
  const int blk   = blockIdx.x;       // sender block
  const int slice = blockIdx.y;       // receiver slice
  const int lane  = tid & 31;
  const int lo    = lane & 15, hi = lane >> 4;
  const int w     = tid >> 5;

  // ---- block-invariant staging ----
  if (tid < 48) sPosS[tid] = pos[blk * 48 + tid];
  if (tid < 32) {
    float s = sigma[tid];
    sSig[tid]   = 1.0f / (s * s);
    sBedge[tid] = bedge[tid];
    sLns[tid]   = lns[tid];
    sLnb[tid]   = lnb[tid];
  }
  if (tid < 64) sMb1[tid] = msgB1[tid];
  if (tid < 128) sWedge[tid] = Wedge[tid];
  for (int idx = tid; idx < 32 * 32; idx += 256) {          // WenvT[n][k]
    int n = idx >> 5, k = idx & 31;
    sWenvT[idx] = f2bf(Wenv[k * 32 + n]);
  }
  for (int idx = tid; idx < 64 * 32; idx += 256) {          // WgT[n][k]
    int n = idx >> 5, k = idx & 31;
    sWgT[idx] = f2bf(Wgate[k * 64 + n]);
  }
  for (int idx = tid; idx < 64 * HIDP; idx += 256) {        // W2T[n][k], pad k>=91
    int n = idx / HIDP, k = idx % HIDP;
    sW2T[idx] = (k < HIDR) ? f2bf(msgW2[k * 64 + n]) : (unsigned short)0;
  }
  {
    const float4* src = (const float4*)(Abuf + blk * 16 * HIDP);
    float4* dst = (float4*)sArows;
    for (int idx = tid; idx < 16 * HIDP / 4; idx += 256) dst[idx] = src[idx];
  }
  __syncthreads();

  float msgacc[2][4] = {{0.f, 0.f, 0.f, 0.f}, {0.f, 0.f, 0.f, 0.f}};

  for (int q = 0; q < TPS; ++q) {
    const int jc = slice * TPS + q;
    const bool diag = (jc == blk);

    // prefetch next tile's B rows into cache (global_prefetch)
    if (q + 1 < TPS && tid < 48)
      __builtin_prefetch(Bbuf + (jc + 1) * 16 * HIDP + tid * 32, 0, 0);

    // ---- stage receiver data (the only workgroup-wide dependence) ----
    if (tid < 48) sPosR[tid] = pos[jc * 48 + tid];
    {
      const float4* src = (const float4*)(Bbuf + jc * 16 * HIDP);
      float4* dst = (float4*)sBrows;
      for (int idx = tid; idx < 16 * HIDP / 4; idx += 256) dst[idx] = src[idx];
    }
    __syncthreads();

    // ---- phase 1: per-edge elementwise (thread e -> row e, wave-local) ----
    const int e  = tid;
    const int il = e >> 4, jl = e & 15;
    float ev0 = sPosS[il * 3 + 0] - sPosR[jl * 3 + 0];
    float ev1 = sPosS[il * 3 + 1] - sPosR[jl * 3 + 1];
    float ev2 = sPosS[il * 3 + 2] - sPosR[jl * 3 + 2];
    float d2 = ev0 * ev0 + ev1 * ev1 + ev2 * ev2;
    float dist = sqrtf(d2);

    float hv[32];
    float mu = 0.f;
#pragma unroll
    for (int k = 0; k < 32; ++k) {
      float pre = ev0 * sWedge[k] + ev1 * sWedge[32 + k] +
                  ev2 * sWedge[64 + k] + dist * sWedge[96 + k] + sBedge[k];
      hv[k] = silu_g(pre);
      mu += hv[k];
    }
    mu *= (1.0f / 32.0f);
    float var = 0.f;
#pragma unroll
    for (int k = 0; k < 32; ++k) { float d = hv[k] - mu; var += d * d; }
    var *= (1.0f / 32.0f);
    float rs = __builtin_amdgcn_rsqf(var + LN_EPS);
#pragma unroll
    for (int k = 0; k < 32; ++k)
      sHn[e * 32 + k] = (hv[k] - mu) * rs * sLns[k] + sLnb[k];   // LN(h)
    {
      unsigned* rrow = (unsigned*)(sRbfA + e * 32);               // RBF A-matrix
#pragma unroll
      for (int k = 0; k < 32; k += 2)
        rrow[k >> 1] = f2bf2(__expf(-d2 * sSig[k]), __expf(-d2 * sSig[k + 1]));
    }
    // no barrier: rows [32w,32w+32) produced and consumed by wave w only

    // ---- phase 2: env = rbf @ Wenv (WMMA); e_embed = LN(h)*env in place ----
#pragma unroll
    for (int m2 = 0; m2 < 2; ++m2) {
      const int m = 2 * w + m2;
      v16bf a = ldsAfrag(sRbfA, 32, 16 * m, 0);
      v8f acc0 = {}, acc1 = {};
      acc0 = wmma_bf16(a, ldsBfrag(sWenvT, 32, 0, 0), acc0);
      acc1 = wmma_bf16(a, ldsBfrag(sWenvT, 32, 16, 0), acc1);
#pragma unroll
      for (int v = 0; v < 8; ++v) {
        const int er = (16 * m + v + 8 * hi) * 32;
        sEA[er + lo]      = f2bf(sHn[er + lo]      * acc0[v]);
        sEA[er + 16 + lo] = f2bf(sHn[er + 16 + lo] * acc1[v]);
      }
    }

    // ---- phase 3: gate = e_embed @ Wgate ; msg = (hid @ W2 + b1) * gate ----
#pragma unroll
    for (int m2 = 0; m2 < 2; ++m2) {
      const int m = 2 * w + m2;                  // sender i_local == m
      v16bf af0 = hidAfrag(sArows, sBrows, m, 0);
      v16bf af1 = hidAfrag(sArows, sBrows, m, 32);
      v16bf af2 = hidAfrag(sArows, sBrows, m, 64);
      v16bf ae  = ldsAfrag(sEA, 32, 16 * m, 0);
#pragma unroll
      for (int nt = 0; nt < 4; ++nt) {
        v8f g = {};
        g = wmma_bf16(ae, ldsBfrag(sWgT, 32, nt * 16, 0), g);
        v8f mm = {};
        mm = wmma_bf16(af0, ldsBfrag(sW2T, HIDP, nt * 16, 0),  mm);
        mm = wmma_bf16(af1, ldsBfrag(sW2T, HIDP, nt * 16, 32), mm);
        mm = wmma_bf16(af2, ldsBfrag(sW2T, HIDP, nt * 16, 64), mm);
        float bb = sMb1[nt * 16 + lo];
        float s = 0.f;
#pragma unroll
        for (int v = 0; v < 8; ++v) {
          int row = v + 8 * hi;                  // receiver j_local
          float val = (mm[v] + bb) * g[v];
          if (diag && row == m) val = 0.f;       // mask self-edge
          s += val;
        }
        s += __shfl_xor(s, 16);                  // fold lane halves (wave32)
        msgacc[m2][nt] += s;
      }
    }
    __syncthreads();                             // protect sBrows/sPosR restage
  }

  // ---- write partial messages (one owner per (sender, feature)) ----
  if (lane < 16) {
#pragma unroll
    for (int m2 = 0; m2 < 2; ++m2) {
      const int i = blk * 16 + 2 * w + m2;
#pragma unroll
      for (int nt = 0; nt < 4; ++nt)
        msg_part[(size_t)slice * NN * 64 + i * 64 + nt * 16 + lane] = msgacc[m2][nt];
    }
  }
}

// ---------------------------------------------------------------------------
// K3: reduce partial messages in fixed order, divide by degree (767)
__global__ void gnn_msgred_k(const float* __restrict__ msg_part,
                             float* __restrict__ msg) {
  int idx = blockIdx.x * blockDim.x + threadIdx.x;
  if (idx >= NN * 64) return;
  float s = 0.f;
  for (int p = 0; p < JS; ++p) s += msg_part[(size_t)p * NN * 64 + idx];
  msg[idx] = s * INV_DEG;
}

// K4: update MLP  concat(H, msg)[128] -> 91 -> 64, residual/sqrt(2)
__global__ __launch_bounds__(128)
void gnn_upd_k(const float* __restrict__ H, const float* __restrict__ msg,
               const float* __restrict__ W0, const float* __restrict__ b0,
               const float* __restrict__ W1, const float* __restrict__ b1,
               float* __restrict__ Hout) {
  __shared__ float sIn[128];
  __shared__ float hh[HIDR];
  const int n = blockIdx.x, t = threadIdx.x;
  sIn[t] = (t < 64) ? H[n * 64 + t] : msg[n * 64 + (t - 64)];
  __syncthreads();
  if (t < HIDR) {
    float s = b0[t];
    for (int k = 0; k < 128; ++k) s += sIn[k] * W0[k * HIDR + t];
    hh[t] = silu_g(s);
  }
  __syncthreads();
  if (t < 64) {
    float u = b1[t];
    for (int k = 0; k < HIDR; ++k) u += hh[k] * W1[k * 64 + t];
    Hout[n * 64 + t] = (sIn[t] + u) * INV_SQRT2;
  }
}

// K5: readout  agg[192] -> 111 -> 64 (no final bias) + embed[charge]
__global__ __launch_bounds__(192)
void gnn_out_k(const float* __restrict__ H0, const float* __restrict__ H1,
               const float* __restrict__ H2, const int* __restrict__ charges,
               const float* __restrict__ nembed,
               const float* __restrict__ W0, const float* __restrict__ b0,
               const float* __restrict__ W1, float* __restrict__ out) {
  __shared__ float agg[192];
  __shared__ float hh[111];
  const int n = blockIdx.x, t = threadIdx.x;
  const float* src = (t < 64) ? H0 : (t < 128) ? H1 : H2;
  agg[t] = src[n * 64 + (t & 63)];
  __syncthreads();
  if (t < 111) {
    float s = b0[t];
    for (int k = 0; k < 192; ++k) s += agg[k] * W0[k * 111 + t];
    hh[t] = silu_g(s);
  }
  __syncthreads();
  if (t < 64) {
    float o = nembed[charges[n] * 64 + t];
    for (int k = 0; k < 111; ++k) o += hh[k] * W1[k * 64 + t];
    out[n * 64 + t] = o;
  }
}

// ---------------------------------------------------------------------------
extern "C" void kernel_launch(void* const* d_in, const int* in_sizes, int n_in,
                              void* d_out, int out_size, void* d_ws, size_t ws_size,
                              hipStream_t stream) {
  (void)in_sizes; (void)n_in; (void)out_size; (void)ws_size;
  // setup_inputs() dict order, DFS, None biases dropped:
  const float* nuclei  = (const float*)d_in[0];
  const int*   charges = (const int*)d_in[1];
  // d_in[2]=senders, d_in[3]=receivers: dense all-pairs graph, structure exploited.
  const float* sigma = (const float*)d_in[4];
  const float* Wenv  = (const float*)d_in[5];
  const float* Wedge = (const float*)d_in[6];
  const float* bedge = (const float*)d_in[7];
  const float* lnsc  = (const float*)d_in[8];
  const float* lnbi  = (const float*)d_in[9];
  const float* embed = (const float*)d_in[10];
  // per layer (stride 9): msgW0, msgW1, msgB0, msgB1, Wgate, updW0, updW1, updB0, updB1
  const int bases[2] = {11, 20};
  const float* nodeEmb = (const float*)d_in[29];
  const float* outW0   = (const float*)d_in[30];
  const float* outW1   = (const float*)d_in[31];
  const float* outB0   = (const float*)d_in[32];

  float* ws = (float*)d_ws;
  float* H0       = ws;                       // 49152
  float* H1       = ws + 49152;
  float* H2       = ws + 98304;
  float* Abuf     = ws + 147456;              // 73728
  float* Bbuf     = ws + 221184;              // 73728
  float* msg      = ws + 294912;              // 49152
  float* msg_part = ws + 344064;              // JS*49152

  gnn_embed_k<<<(NN * 64 + 255) / 256, 256, 0, stream>>>(charges, embed, H0);

  const float* Hin = H0;
  float* Houts[2] = {H1, H2};
  const int bases_arr[2] = {bases[0], bases[1]};
  for (int l = 0; l < 2; ++l) {
    const int b = bases_arr[l];
    const float* msgW0 = (const float*)d_in[b + 0];
    const float* msgW1 = (const float*)d_in[b + 1];
    const float* msgB0 = (const float*)d_in[b + 2];
    const float* msgB1 = (const float*)d_in[b + 3];
    const float* Wgate = (const float*)d_in[b + 4];
    const float* updW0 = (const float*)d_in[b + 5];
    const float* updW1 = (const float*)d_in[b + 6];
    const float* updB0 = (const float*)d_in[b + 7];
    const float* updB1 = (const float*)d_in[b + 8];

    gnn_precab_k<<<(NN * HIDP + 255) / 256, 256, 0, stream>>>(Hin, msgW0, msgB0,
                                                              Abuf, Bbuf);
    dim3 grid(NB, JS);
    gnn_edge_k<<<grid, 256, SMEM_BYTES, stream>>>(nuclei, Abuf, Bbuf, sigma, Wenv,
                                                  Wedge, bedge, lnsc, lnbi, Wgate,
                                                  msgW1, msgB1, msg_part);
    gnn_msgred_k<<<(NN * 64 + 255) / 256, 256, 0, stream>>>(msg_part, msg);
    gnn_upd_k<<<NN, 128, 0, stream>>>(Hin, msg, updW0, updB0, updW1, updB1,
                                      Houts[l]);
    Hin = Houts[l];
  }

  gnn_out_k<<<NN, 192, 0, stream>>>(H0, H1, H2, charges, nodeEmb,
                                    outW0, outB0, outW1, (float*)d_out);
}
